// SpatioTemporalTransformerGNN_12412455485636
// MI455X (gfx1250) — matrix operand order repaired and checked by reading, not compiled, not run on hardware
//
#include <hip/hip_runtime.h>

typedef __attribute__((ext_vector_type(16))) _Float16 v16h;
typedef __attribute__((ext_vector_type(8)))  _Float16 v8h;
typedef __attribute__((ext_vector_type(8)))  float    v8f;

#define NN    20000
#define EE    320000
#define SS    4
#define FIN   32
#define HIDW  64
#define NHEAD 4
#define EMBW  8

// ---------------- helpers ----------------
__device__ __forceinline__ unsigned fkey(float f) {
  unsigned u = __float_as_uint(f);
  return (u & 0x80000000u) ? ~u : (u | 0x80000000u);
}
__device__ __forceinline__ float fdec(unsigned k) {
  unsigned u = (k & 0x80000000u) ? (k & 0x7fffffffu) : ~k;
  return __uint_as_float(u);
}
__device__ __forceinline__ float elu1(float v) {
  return v > 0.f ? v : (__expf(v) - 1.f);
}

// ---------------- weight prep (f32 -> f16, transposed to out x Kpad) ----------------
__global__ void prep_w1(const float* __restrict__ W1l, const float* __restrict__ b1l,
                        const float* __restrict__ W1r, const float* __restrict__ b1r,
                        _Float16* __restrict__ BT, float* __restrict__ bc) {
  int i = blockIdx.x * blockDim.x + threadIdx.x;      // 512*64
  if (i >= 512 * 64) return;
  int o = i >> 6, k = i & 63;
  float v = 0.f;
  if (k < 40) v = (o < 256) ? W1l[k * 256 + o] : W1r[k * 256 + (o - 256)];
  BT[i] = (_Float16)v;
  if (k == 0) bc[o] = (o < 256) ? b1l[o] : b1r[o - 256];
}

__global__ void prep_w2(const float* __restrict__ W2l, const float* __restrict__ b2l,
                        const float* __restrict__ W2r, const float* __restrict__ b2r,
                        _Float16* __restrict__ BT, float* __restrict__ bc) {
  int i = blockIdx.x * blockDim.x + threadIdx.x;      // 128*256
  if (i >= 128 * 256) return;
  int o = i >> 8, k = i & 255;
  float v = (o < 64) ? W2l[k * 64 + o] : W2r[k * 64 + (o - 64)];
  BT[i] = (_Float16)v;
  if (k == 0) bc[o] = (o < 64) ? b2l[o] : b2r[o - 64];
}

__global__ void prep_wq(const float* __restrict__ Wqkv, _Float16* __restrict__ BT) {
  int i = blockIdx.x * blockDim.x + threadIdx.x;      // 192*64  (already out x in)
  if (i >= 192 * 64) return;
  BT[i] = (_Float16)Wqkv[i];
}

// ---------------- feature concat (x || emb[aq]) padded to 64, f16 ----------------
__global__ void concat_xa(const float* __restrict__ x, const int* __restrict__ aq,
                          const float* __restrict__ emb, _Float16* __restrict__ xa,
                          int t) {
  int i = blockIdx.x * blockDim.x + threadIdx.x;      // NN*64
  if (i >= NN * 64) return;
  int n = i >> 6, c = i & 63;
  float v = 0.f;
  if (c < FIN)             v = x[((size_t)n * SS + t) * FIN + c];
  else if (c < FIN + EMBW) v = emb[aq[n] * EMBW + (c - FIN)];
  xa[i] = (_Float16)v;
}

// ---------------- WMMA GEMM: C(MxNc) = A16(MxK) * BT16(NcxK)^T + bias ----------------
// One block (8 waves) computes a 128x64 output tile; the 64xK f16 B panel is staged
// into LDS once per block via async global->LDS copies (ASYNCcnt), then each wave
// runs 16x64 of output with v_wmma_f32_16x16x32_f16, B fragments from ds_load_b128.
// K in {64,256} -> staging loop trip counts are wave-uniform, EXEC stays all-ones.
__global__ void __launch_bounds__(256)
wmma_gemm(const _Float16* __restrict__ A, const _Float16* __restrict__ BT,
          const float* __restrict__ bias, float* __restrict__ C,
          int M, int K, int Nc) {
  __shared__ _Float16 bpan[64 * 256];                 // 32 KB max (K <= 256)

  const int tilesN = Nc >> 6;
  const int btm = blockIdx.x / tilesN;                // 128-row block index
  const int tn  = blockIdx.x - btm * tilesN;

  // ---- async-stage B panel: rows [tn*64, tn*64+64) x K halves ----
  {
    const int chunks = (64 * K) >> 3;                 // 16-byte chunks
    const _Float16* gbase = BT + (size_t)tn * 64 * K;
    for (int it = threadIdx.x; it < chunks; it += 256) {
      unsigned lds_off = (unsigned)(unsigned long long)(bpan + it * 8);
      const _Float16* gp = gbase + it * 8;
      asm volatile("global_load_async_to_lds_b128 %0, %1, off"
                   :: "v"(lds_off), "v"(gp) : "memory");
    }
    asm volatile("s_wait_asynccnt 0x0" ::: "memory");
  }
  __syncthreads();

  const int lane = threadIdx.x & 31;
  const int wv   = threadIdx.x >> 5;
  const int tm   = btm * 8 + wv;
  if (tm * 16 >= M) return;                           // after the only barrier
  const int hi  = lane >> 4;
  const int ln  = lane & 15;
  const int row = tm * 16 + ln;

  v8f acc[4] = {};
  for (int k0 = 0; k0 < K; k0 += 32) {
    // A fragment: lane holds row `row`, K = k0+8*hi+[0..7] and k0+8*hi+16+[0..7]
    const _Float16* ap = A + (size_t)row * K + k0 + hi * 8;
    v8h alo = *(const v8h*)ap;
    v8h ahi = *(const v8h*)(ap + 16);
    v16h a;
#pragma unroll
    for (int j = 0; j < 8; ++j) { a[j] = alo[j]; a[8 + j] = ahi[j]; }

    const int kb2 = k0 + hi * 8;
#pragma unroll
    for (int i = 0; i < 4; ++i) {
      // B fragment from LDS: panel row i*16+ln, K = kb2+[0..7] and kb2+16+[0..7]
      const _Float16* bp = bpan + (i * 16 + ln) * K + kb2;
      v8h b0 = *(const v8h*)bp;
      v8h b1 = *(const v8h*)(bp + 16);
      v16h b;
#pragma unroll
      for (int j = 0; j < 8; ++j) { b[j] = b0[j]; b[8 + j] = b1[j]; }
      acc[i] = __builtin_amdgcn_wmma_f32_16x16x32_f16(
          false, a, false, b, (short)0, acc[i], false, false);
    }
  }
#pragma unroll
  for (int i = 0; i < 4; ++i) {
    int colc = tn * 64 + i * 16 + ln;
    float bv = bias ? bias[colc] : 0.f;
#pragma unroll
    for (int r = 0; r < 8; ++r) {
      int rowc = tm * 16 + hi * 8 + r;                // C layout: VGPR r <-> M = r + 8*hi
      if (rowc < M) C[(size_t)rowc * Nc + colc] = acc[i][r] + bv;
    }
  }
}

// ---------------- GATv2 edge kernels (generic over H, C) ----------------
__global__ void edge_score(const int* __restrict__ src, const int* __restrict__ dst,
                           const float* __restrict__ xlr, int ldx, int xl_off, int xr_off,
                           const float* __restrict__ att, int H, int C,
                           float* __restrict__ ebuf, unsigned* __restrict__ emax,
                           int total) {
  int i = blockIdx.x * blockDim.x + threadIdx.x;
  if (i >= total) return;
  int e = i / H, h = i - e * H;
  int s = src[e], d = dst[e];
  const float* pl = xlr + (size_t)s * ldx + xl_off + h * C;
  const float* pr = xlr + (size_t)d * ldx + xr_off + h * C;
  const float* pa = att + h * C;
  float acc = 0.f;
  for (int c = 0; c < C; ++c) {
    float v = pl[c] + pr[c];
    v = v > 0.f ? v : 0.2f * v;             // leaky_relu(0.2)
    acc += pa[c] * v;
  }
  ebuf[i] = acc;
  atomicMax(&emax[(size_t)d * H + h], fkey(acc));
}

__global__ void edge_w(const int* __restrict__ dst, float* __restrict__ ebuf,
                       const unsigned* __restrict__ emax, float* __restrict__ denom,
                       int H, int total) {
  int i = blockIdx.x * blockDim.x + threadIdx.x;
  if (i >= total) return;
  int e = i / H, h = i - e * H;
  int d = dst[e];
  float m = fdec(emax[(size_t)d * H + h]);
  float w = __expf(ebuf[i] - m);
  ebuf[i] = w;
  atomicAdd(&denom[(size_t)d * H + h], w);
}

__global__ void edge_scatter(const int* __restrict__ src, const int* __restrict__ dst,
                             const float* __restrict__ wbuf, const float* __restrict__ denom,
                             const float* __restrict__ xlr, int ldx, int xl_off,
                             int H, int C, float* __restrict__ accum, int total) {
  int i = blockIdx.x * blockDim.x + threadIdx.x;
  if (i >= total) return;
  int e = i / H, h = i - e * H;
  int s = src[e], d = dst[e];
  float den = denom[(size_t)d * H + h];
  float alpha = wbuf[i] / fmaxf(den, 1e-16f);
  const float* pl = xlr + (size_t)s * ldx + xl_off + h * C;
  float* po = accum + ((size_t)d * H + h) * C;
  for (int c = 0; c < C; ++c) atomicAdd(&po[c], alpha * pl[c]);
}

// ---------------- activations / converts ----------------
__global__ void h1_act(const float* __restrict__ accum, const float* __restrict__ bias,
                       _Float16* __restrict__ h16) {
  int i = blockIdx.x * blockDim.x + threadIdx.x;      // NN*256
  if (i >= NN * 256) return;
  h16[i] = (_Float16)elu1(accum[i] + bias[i & 255]);
}

__global__ void seq_write(const float* __restrict__ accum, const float* __restrict__ bias,
                          float* __restrict__ seq, int t) {
  int i = blockIdx.x * blockDim.x + threadIdx.x;      // NN*64
  if (i >= NN * 64) return;
  int n = i >> 6, c = i & 63;
  seq[((size_t)n * SS + t) * 64 + c] = elu1(accum[i] + bias[c]);
}

__global__ void cvt16(const float* __restrict__ srcf, _Float16* __restrict__ dsth, int n) {
  int i = blockIdx.x * blockDim.x + threadIdx.x;
  if (i < n) dsth[i] = (_Float16)srcf[i];
}

// ---------------- fused temporal attention (s = S-1 only) + LN + MLP ----------------
// one wave per node; lane l owns channels {l, l+32}
__global__ void __launch_bounds__(256)
temporal_fuse(const float* __restrict__ qkv,  // (N*S, 192) q|k|v
              const float* __restrict__ seq,  // (N, S, 64)
              const float* __restrict__ Wo, const float* __restrict__ bo,
              const float* __restrict__ gamma, const float* __restrict__ beta,
              const float* __restrict__ Wf1, const float* __restrict__ bf1,
              const float* __restrict__ Wout, const float* __restrict__ bout,
              float* __restrict__ out) {
  __shared__ float sm[8][64];
  __shared__ float sscore[8][4][4];           // [wave][t][head]
  const int wv = threadIdx.x >> 5;
  const int lane = threadIdx.x & 31;
  int n = blockIdx.x * 8 + wv;
  const bool active = n < NN;
  if (n >= NN) n = NN - 1;                    // keep all waves in barriers
  const int c0 = lane, c1 = lane + 32;

  // scores for query s=3 (Dh=16, scale 1/4)
  const float* q3 = qkv + ((size_t)n * SS + 3) * 192;
  float q0 = q3[c0], q1 = q3[c1];
#pragma unroll
  for (int t = 0; t < 4; ++t) {
    const float* kt = qkv + ((size_t)n * SS + t) * 192 + 64;
    float p0 = q0 * kt[c0], p1 = q1 * kt[c1];
#pragma unroll
    for (int m = 1; m < 16; m <<= 1) {
      p0 += __shfl_xor(p0, m, 32);
      p1 += __shfl_xor(p1, m, 32);
    }
    if ((lane & 15) == 0) {
      int g = lane >> 4;
      sscore[wv][t][g]     = p0 * 0.25f;
      sscore[wv][t][2 + g] = p1 * 0.25f;
    }
  }
  __syncthreads();

  const int h0 = lane >> 4, h1 = 2 + (lane >> 4);
  float a0[4], a1[4];
  {
    float m0 = -1e30f, m1 = -1e30f;
#pragma unroll
    for (int t = 0; t < 4; ++t) {
      m0 = fmaxf(m0, sscore[wv][t][h0]);
      m1 = fmaxf(m1, sscore[wv][t][h1]);
    }
    float s0 = 0.f, s1 = 0.f;
#pragma unroll
    for (int t = 0; t < 4; ++t) {
      a0[t] = __expf(sscore[wv][t][h0] - m0); s0 += a0[t];
      a1[t] = __expf(sscore[wv][t][h1] - m1); s1 += a1[t];
    }
#pragma unroll
    for (int t = 0; t < 4; ++t) { a0[t] /= s0; a1[t] /= s1; }
  }
  float o0 = 0.f, o1 = 0.f;
#pragma unroll
  for (int t = 0; t < 4; ++t) {
    const float* vt = qkv + ((size_t)n * SS + t) * 192 + 128;
    o0 += a0[t] * vt[c0];
    o1 += a1[t] * vt[c1];
  }
  __syncthreads();
  sm[wv][c0] = o0; sm[wv][c1] = o1;
  __syncthreads();

  // o @ Wo^T + bo + residual
  float y0 = bo[c0], y1 = bo[c1];
  for (int k = 0; k < 64; ++k) {
    float ov = sm[wv][k];
    y0 += Wo[c0 * 64 + k] * ov;
    y1 += Wo[c1 * 64 + k] * ov;
  }
  const float* sq = seq + ((size_t)n * SS + 3) * 64;
  y0 += sq[c0]; y1 += sq[c1];

  // LayerNorm over 64
  float s = y0 + y1, ss2 = y0 * y0 + y1 * y1;
#pragma unroll
  for (int m = 1; m < 32; m <<= 1) {
    s   += __shfl_xor(s, m, 32);
    ss2 += __shfl_xor(ss2, m, 32);
  }
  float mu  = s * (1.f / 64.f);
  float var = ss2 * (1.f / 64.f) - mu * mu;
  float inv = rsqrtf(var + 1e-5f);
  y0 = (y0 - mu) * inv * gamma[c0] + beta[c0];
  y1 = (y1 - mu) * inv * gamma[c1] + beta[c1];

  __syncthreads();
  sm[wv][c0] = y0; sm[wv][c1] = y1;
  __syncthreads();

  // MLP: relu(y @ Wf1 + bf1) @ Wout + bout ; lane j owns hidden unit j
  float f = bf1[lane];
  for (int c = 0; c < 64; ++c) f += sm[wv][c] * Wf1[c * 32 + lane];
  f = fmaxf(f, 0.f);
#pragma unroll
  for (int k = 0; k < 3; ++k) {
    float p = f * Wout[lane * 3 + k];
#pragma unroll
    for (int m = 1; m < 32; m <<= 1) p += __shfl_xor(p, m, 32);
    if (active && lane == 0) out[(size_t)n * 3 + k] = p + bout[k];
  }
}

// ---------------- host ----------------
static inline int ceil_div(long long a, long long b) { return (int)((a + b - 1) / b); }

extern "C" void kernel_launch(void* const* d_in, const int* in_sizes, int n_in,
                              void* d_out, int out_size, void* d_ws, size_t ws_size,
                              hipStream_t stream) {
  (void)in_sizes; (void)n_in; (void)out_size; (void)ws_size;
  const float* x     = (const float*)d_in[0];
  const int*   ei    = (const int*)d_in[1];
  const int*   aq    = (const int*)d_in[2];
  const float* emb   = (const float*)d_in[3];
  const float* W1l   = (const float*)d_in[4];
  const float* b1l   = (const float*)d_in[5];
  const float* W1r   = (const float*)d_in[6];
  const float* b1r   = (const float*)d_in[7];
  const float* att1  = (const float*)d_in[8];
  const float* bias1 = (const float*)d_in[9];
  const float* W2l   = (const float*)d_in[10];
  const float* b2l   = (const float*)d_in[11];
  const float* W2r   = (const float*)d_in[12];
  const float* b2r   = (const float*)d_in[13];
  const float* att2  = (const float*)d_in[14];
  const float* bias2 = (const float*)d_in[15];
  const float* Wqkv  = (const float*)d_in[16];
  const float* bqkv  = (const float*)d_in[17];
  const float* Wo    = (const float*)d_in[18];
  const float* bo    = (const float*)d_in[19];
  const float* gamma = (const float*)d_in[20];
  const float* beta  = (const float*)d_in[21];
  const float* Wf1   = (const float*)d_in[22];
  const float* bf1   = (const float*)d_in[23];
  const float* Wout  = (const float*)d_in[24];
  const float* bout  = (const float*)d_in[25];
  const int* src = ei;
  const int* dst = ei + EE;

  char* ws = (char*)d_ws;
  size_t off = 0;
  auto alloc = [&](size_t bytes) { size_t o = off; off += (bytes + 255) & ~(size_t)255; return o; };

  size_t o_bt1    = alloc(512 * 64 * 2);
  size_t o_bc1    = alloc(512 * 4);
  size_t o_bt2    = alloc(128 * 256 * 2);
  size_t o_bc2    = alloc(128 * 4);
  size_t o_btq    = alloc(192 * 64 * 2);
  size_t o_xa16   = alloc((size_t)NN * 64 * 2);
  size_t o_out1lr = alloc((size_t)NN * 512 * 4);      // phase-A; aliased by qkv later
  size_t o_accum1 = alloc((size_t)NN * 256 * 4);      // contiguous with out1lr (both 256B-mult)
  size_t o_e1     = alloc((size_t)EE * 4 * 4);
  size_t o_emax1  = alloc((size_t)NN * 4 * 4);
  size_t o_den1   = alloc((size_t)NN * 4 * 4);
  size_t o_h116   = alloc((size_t)NN * 256 * 2);      // aliased by seq16 later
  size_t o_out2lr = alloc((size_t)NN * 128 * 4);
  size_t o_e2     = alloc((size_t)EE * 4);
  size_t o_emax2  = alloc((size_t)NN * 4);
  size_t o_den2   = alloc((size_t)NN * 4);
  size_t o_seq    = alloc((size_t)NN * SS * 64 * 4);
  size_t o_qkv    = o_out1lr;                          // N*S*192*4 == N*(512+256)*4
  size_t o_seq16  = o_h116;

  _Float16* bt1    = (_Float16*)(ws + o_bt1);
  float*    bc1    = (float*)(ws + o_bc1);
  _Float16* bt2    = (_Float16*)(ws + o_bt2);
  float*    bc2    = (float*)(ws + o_bc2);
  _Float16* btq    = (_Float16*)(ws + o_btq);
  _Float16* xa16   = (_Float16*)(ws + o_xa16);
  float*    out1lr = (float*)(ws + o_out1lr);
  float*    accum1 = (float*)(ws + o_accum1);
  float*    e1     = (float*)(ws + o_e1);
  unsigned* emax1  = (unsigned*)(ws + o_emax1);
  float*    den1   = (float*)(ws + o_den1);
  _Float16* h116   = (_Float16*)(ws + o_h116);
  float*    out2lr = (float*)(ws + o_out2lr);
  float*    e2     = (float*)(ws + o_e2);
  unsigned* emax2  = (unsigned*)(ws + o_emax2);
  float*    den2   = (float*)(ws + o_den2);
  float*    seq    = (float*)(ws + o_seq);
  float*    qkv    = (float*)(ws + o_qkv);
  _Float16* seq16  = (_Float16*)(ws + o_seq16);

  // weight prep
  prep_w1<<<ceil_div(512 * 64, 256), 256, 0, stream>>>(W1l, b1l, W1r, b1r, bt1, bc1);
  prep_w2<<<ceil_div(128 * 256, 256), 256, 0, stream>>>(W2l, b2l, W2r, b2r, bt2, bc2);
  prep_wq<<<ceil_div(192 * 64, 256), 256, 0, stream>>>(Wqkv, btq);

  const int EH1 = EE * NHEAD, EH2 = EE;

  for (int t = 0; t < SS; ++t) {
    concat_xa<<<ceil_div((long long)NN * 64, 256), 256, 0, stream>>>(x, aq, emb, xa16, t);

    // layer-1 L/R projections: (N,64)x(64,512); grid = tilesM8 * tilesN
    { int grid = ceil_div(NN, 128) * (512 / 64);
      wmma_gemm<<<grid, 256, 0, stream>>>(xa16, bt1, bc1, out1lr, NN, 64, 512); }

    hipMemsetAsync(accum1, 0, (size_t)NN * 256 * 4, stream);
    hipMemsetAsync(emax1, 0, (size_t)NN * 4 * 4, stream);
    hipMemsetAsync(den1, 0, (size_t)NN * 4 * 4, stream);

    edge_score<<<ceil_div(EH1, 256), 256, 0, stream>>>(src, dst, out1lr, 512, 0, 256,
                                                       att1, NHEAD, HIDW, e1, emax1, EH1);
    edge_w<<<ceil_div(EH1, 256), 256, 0, stream>>>(dst, e1, emax1, den1, NHEAD, EH1);
    edge_scatter<<<ceil_div(EH1, 256), 256, 0, stream>>>(src, dst, e1, den1, out1lr, 512, 0,
                                                         NHEAD, HIDW, accum1, EH1);
    h1_act<<<ceil_div((long long)NN * 256, 256), 256, 0, stream>>>(accum1, bias1, h116);

    // layer-2 L/R projections: (N,256)x(256,128)
    { int grid = ceil_div(NN, 128) * (128 / 64);
      wmma_gemm<<<grid, 256, 0, stream>>>(h116, bt2, bc2, out2lr, NN, 256, 128); }

    // reuse accum1's first N*64 floats for layer-2 accumulation
    float* accum2 = accum1;
    hipMemsetAsync(accum2, 0, (size_t)NN * 64 * 4, stream);
    hipMemsetAsync(emax2, 0, (size_t)NN * 4, stream);
    hipMemsetAsync(den2, 0, (size_t)NN * 4, stream);

    edge_score<<<ceil_div(EH2, 256), 256, 0, stream>>>(src, dst, out2lr, 128, 0, 64,
                                                       att2, 1, HIDW, e2, emax2, EH2);
    edge_w<<<ceil_div(EH2, 256), 256, 0, stream>>>(dst, e2, emax2, den2, 1, EH2);
    edge_scatter<<<ceil_div(EH2, 256), 256, 0, stream>>>(src, dst, e2, den2, out2lr, 128, 0,
                                                         1, HIDW, accum2, EH2);
    seq_write<<<ceil_div((long long)NN * 64, 256), 256, 0, stream>>>(accum2, bias2, seq, t);
  }

  // QKV projection: (N*S,64) x (64,192)   (qkv aliases the dead phase-A region)
  cvt16<<<ceil_div((long long)NN * SS * 64, 256), 256, 0, stream>>>(seq, seq16, NN * SS * 64);
  { int grid = ceil_div(NN * SS, 128) * (192 / 64);
    wmma_gemm<<<grid, 256, 0, stream>>>(seq16, btq, bqkv, qkv, NN * SS, 64, 192); }

  // fused attention + Wo + residual + LayerNorm + MLP -> (N,3)
  temporal_fuse<<<ceil_div(NN, 8), 256, 0, stream>>>(qkv, seq, Wo, bo, gamma, beta,
                                                     Wf1, bf1, Wout, bout, (float*)d_out);
}
